// MultiSwinTransformerBlock_15762529976476
// MI455X (gfx1250) — matrix operand order repaired and checked by reading, not compiled, not run on hardware
//
#include <hip/hip_runtime.h>
#include <math.h>

// ---------------------------------------------------------------------------
// MultiSwinTransformerBlock for MI455X (gfx1250), bf16 WMMA implementation.
// B=16, H=W=56, C=192, NH=6, WS=7 -> Bw=1024 windows, NW=49, HD=32.
// Big GEMM-shaped work (QKV, window convs as implicit GEMM, MLP) runs on
// v_wmma_f32_16x16x32_bf16; LDS tiles are filled with the CDNA5 async
// global->LDS path (global_load_async_to_lds_b128 + s_wait_asynccnt).
// Attention (only ~4 GF) runs on VALU with LDS.
// ---------------------------------------------------------------------------

typedef __attribute__((ext_vector_type(16))) __bf16 v16bf;
typedef __attribute__((ext_vector_type(8)))  float  v8f;

#define TT   50176      // total token rows (Bw*NW = B*N)
#define CC   192
#define BW   1024
#define NWIN 49
#define SCALE_Q 0.17677669529663687f  // 32^-0.5

struct BF16x16 { uint4 lo, hi; };

// A-matrix fragment (16x32 bf16, M x K): lane -> M = lane&15,
// k elements: {0..7,16..23} for lanes 0-15, +8 for lanes 16-31.
__device__ __forceinline__ v16bf frag_a(const __bf16* row, int lane) {
  const int a0 = (lane >= 16) ? 8 : 0;
  BF16x16 p;
  p.lo = *(const uint4*)(row + a0);
  p.hi = *(const uint4*)(row + a0 + 16);
  return __builtin_bit_cast(v16bf, p);
}
// B-matrix fragment (32x16 bf16, K x N) stored as [n][k] rows: lane -> N =
// lane&15, k elements: {0..7,16..23} for lanes 0-15, +8 for lanes 16-31.
__device__ __forceinline__ v16bf frag_b(const __bf16* row, int lane) {
  const int b0 = (lane >= 16) ? 8 : 0;
  BF16x16 p;
  p.lo = *(const uint4*)(row + b0);
  p.hi = *(const uint4*)(row + b0 + 16);
  return __builtin_bit_cast(v16bf, p);
}

// CDNA5 async global -> LDS copy, GVS mode: SADDR(sgpr64) + voff(vgpr32).
// 16 bytes per active lane; tracked by ASYNCcnt.
__device__ __forceinline__ void async_g2l_b128(void* lds, const void* sbase,
                                               unsigned voff) {
  unsigned l = (unsigned)(uintptr_t)lds;   // low 32 bits = LDS byte offset
  asm volatile("global_load_async_to_lds_b128 %0, %1, %2"
               :: "v"(l), "v"(voff), "s"(sbase)
               : "memory");
}
__device__ __forceinline__ void wait_async0() {
  asm volatile("s_wait_asynccnt 0x0" ::: "memory");
}

__device__ __forceinline__ float gelu_f(float x) {
  return 0.5f * x * (1.0f + erff(x * 0.7071067811865475f));
}

// ---------------------------------------------------------------------------
// K0: relative-position-bias table: rpb[h][i][j], h<6, i,j<49
// ---------------------------------------------------------------------------
__global__ void rpb_kernel(const float* __restrict__ w1, const float* __restrict__ b1,
                           const float* __restrict__ w2, const float* __restrict__ b2,
                           float* __restrict__ rpb) {
  int idx = blockIdx.x * 64 + threadIdx.x;
  if (idx >= NWIN * NWIN) return;
  int i = idx / NWIN, j = idx % NWIN;
  float dy = (float)(i / 7 - j / 7);
  float dx = (float)(i % 7 - j % 7);
  float sy = (dy > 0.f) ? 1.f : ((dy < 0.f) ? -1.f : 0.f);
  float sx = (dx > 0.f) ? 1.f : ((dx < 0.f) ? -1.f : 0.f);
  float ry = sy * log1pf(fabsf(dy));
  float rx = sx * log1pf(fabsf(dx));
  float acc[6] = {0.f, 0.f, 0.f, 0.f, 0.f, 0.f};
  for (int u = 0; u < 256; ++u) {
    float hv = fmaxf(w1[u * 2 + 0] * ry + w1[u * 2 + 1] * rx + b1[u], 0.f);
#pragma unroll
    for (int h = 0; h < 6; ++h) acc[h] += hv * w2[h * 256 + u];
  }
  for (int h = 0; h < 6; ++h) rpb[h * (NWIN * NWIN) + idx] = acc[h] + b2[h];
}

// ---------------------------------------------------------------------------
// K1: window-partition + f32 -> bf16 for the three streams
// ---------------------------------------------------------------------------
__global__ __launch_bounds__(256) void prep_kernel(
    const float* __restrict__ x, const float* __restrict__ a, const float* __restrict__ d,
    __bf16* __restrict__ xw, __bf16* __restrict__ aw, __bf16* __restrict__ dw) {
  size_t idx = (size_t)blockIdx.x * 256 + threadIdx.x;   // < TT*CC
  int c = (int)(idx % CC);
  int g = (int)(idx / CC);
  int win = g / NWIN, t = g % NWIN;
  int b  = win >> 6;
  int wh = (win >> 3) & 7, ww = win & 7;
  int th = t / 7, tw = t % 7;
  int n = (wh * 7 + th) * 56 + (ww * 7 + tw);
  size_t src = ((size_t)b * 3136 + n) * CC + c;
  xw[idx] = (__bf16)x[src];
  aw[idx] = (__bf16)a[src];
  dw[idx] = (__bf16)d[src];
}

// ---------------------------------------------------------------------------
// K2: flat f32 -> bf16 convert (weights)
// ---------------------------------------------------------------------------
__global__ void cvt_kernel(const float* __restrict__ s, __bf16* __restrict__ o, int n) {
  int i = blockIdx.x * 256 + threadIdx.x;
  if (i < n) o[i] = (__bf16)s[i];
}

// K3: conv weight repack (Cout,Cin,3,3) f32 -> [tap][cout][cin] bf16
template <int CIN>
__global__ void repack_conv_kernel(const float* __restrict__ w, __bf16* __restrict__ o) {
  int idx = blockIdx.x * 256 + threadIdx.x;   // grid sized exactly
  int ci = idx % CIN;
  int t  = idx / CIN;            // t = tap*192 + co
  int co = t % CC;
  int tap = t / CC;
  o[idx] = (__bf16)w[((size_t)co * CIN + ci) * 9 + tap];
}

// ---------------------------------------------------------------------------
// K4: depthwise 3x3 conv over each 7x7 window (window-local zero padding)
// ---------------------------------------------------------------------------
__global__ __launch_bounds__(256) void dwconv_kernel(
    const __bf16* __restrict__ xw, const float* __restrict__ w,
    const float* __restrict__ b, __bf16* __restrict__ out) {
  size_t idx = (size_t)blockIdx.x * 256 + threadIdx.x;   // < TT*CC
  int c = (int)(idx % CC);
  int g = (int)(idx / CC);
  int win = g / NWIN, t = g % NWIN;
  int py = t / 7, px = t % 7;
  float acc = b[c];
#pragma unroll
  for (int ky = 0; ky < 3; ++ky) {
#pragma unroll
    for (int kx = 0; kx < 3; ++kx) {
      int sy = py + ky - 1, sx = px + kx - 1;
      if (sy >= 0 && sy < 7 && sx >= 0 && sx < 7) {
        acc += (float)xw[((size_t)win * NWIN + sy * 7 + sx) * CC + c] * w[c * 9 + ky * 3 + kx];
      }
    }
  }
  out[idx] = (__bf16)acc;
}

// ---------------------------------------------------------------------------
// K5: generic bf16 WMMA GEMM: out = act(A[MxK] @ B[NxK]^T + bias) (+resid)
// tile 128x64, BK=32, 256 threads = 8 waves laid 4(M) x 2(N), 2x2 wmma/wave.
// LDS tiles filled by async global->LDS (ASYNCcnt), no VGPR round-trip.
// EPI: 0 = store bf16, 1 = GELU + store bf16, 2 = +resid, store f32
// ---------------------------------------------------------------------------
template <int EPI>
__global__ __launch_bounds__(256) void gemm_kernel(
    const __bf16* __restrict__ A, const __bf16* __restrict__ B,
    const float* __restrict__ bias, void* __restrict__ outp,
    const float* __restrict__ resid, int M, int N, int K) {
  __shared__ __align__(16) __bf16 As[128 * 40];
  __shared__ __align__(16) __bf16 Bs[64 * 40];
  const int tid  = threadIdx.x;
  const int lane = tid & 31;
  const int wave = tid >> 5;
  const int wm = wave & 3;          // 32-row granule
  const int wn = wave >> 2;         // 32-col granule
  const int lm = lane & 15;
  const int bm = blockIdx.x * 128;
  const int bn = blockIdx.y * 64;

  v8f acc[2][2] = {};
  const int ar = tid >> 1, ac = (tid & 1) * 16;   // A-tile stage coords
  const int br = tid >> 2, bc = (tid & 3) * 8;    // B-tile stage coords
  const unsigned a_voff0 = (unsigned)(((size_t)(bm + ar) * K + ac) * 2);
  const unsigned b_voff0 = (unsigned)(((size_t)(bn + br) * K + bc) * 2);

  for (int k0 = 0; k0 < K; k0 += 32) {
    {
      unsigned va = a_voff0 + (unsigned)(k0 * 2);
      async_g2l_b128(&As[ar * 40 + ac],     A, va);
      async_g2l_b128(&As[ar * 40 + ac + 8], A, va + 16);
      if (k0 + 32 < K)
        __builtin_prefetch(A + (size_t)(bm + ar) * K + k0 + 32 + ac, 0, 1);
    }
    {
      unsigned vb = b_voff0 + (unsigned)(k0 * 2);
      async_g2l_b128(&Bs[br * 40 + bc], B, vb);
    }
    wait_async0();
    __syncthreads();
#pragma unroll
    for (int mi = 0; mi < 2; ++mi) {
      v16bf af = frag_a(&As[(wm * 32 + mi * 16 + lm) * 40], lane);
#pragma unroll
      for (int ni = 0; ni < 2; ++ni) {
        v16bf bf = frag_b(&Bs[(wn * 32 + ni * 16 + lm) * 40], lane);
        acc[mi][ni] = __builtin_amdgcn_wmma_f32_16x16x32_bf16(
            false, af, false, bf, (short)0, acc[mi][ni], false, false);
      }
    }
    __syncthreads();
  }

  const int hi8 = (lane >= 16) ? 8 : 0;
#pragma unroll
  for (int mi = 0; mi < 2; ++mi) {
#pragma unroll
    for (int ni = 0; ni < 2; ++ni) {
      int n = bn + wn * 32 + ni * 16 + lm;
      float bv = bias[n];
#pragma unroll
      for (int r = 0; r < 8; ++r) {
        size_t m = (size_t)(bm + wm * 32 + mi * 16 + r + hi8);
        float v = acc[mi][ni][r] + bv;
        if (EPI == 0) {
          ((__bf16*)outp)[m * N + n] = (__bf16)v;
        } else if (EPI == 1) {
          ((__bf16*)outp)[m * N + n] = (__bf16)gelu_f(v);
        } else {
          ((float*)outp)[m * N + n] = v + resid[m * N + n];
        }
      }
    }
  }
}

// ---------------------------------------------------------------------------
// K6: cosine attention; one block = (window, head, branch), 64 threads,
// one query row per thread, K/V(+conv) staged in LDS.
// ---------------------------------------------------------------------------
__global__ __launch_bounds__(64) void attn_kernel(
    const __bf16* __restrict__ qkvx, const __bf16* __restrict__ qkva,
    const __bf16* __restrict__ qkvd, const __bf16* __restrict__ convbf,
    const float* __restrict__ tau, const float* __restrict__ rpb,
    __bf16* __restrict__ cat) {
  const int idx = blockIdx.x;
  const int br = idx & 1;
  const int h  = (idx >> 1) % 6;
  const int w  = idx / 12;
  const __bf16* qkvbr = br ? qkvd : qkva;

  __shared__ float ks[NWIN][32];
  __shared__ float vs[NWIN][32];
  __shared__ float kn[NWIN];
  __shared__ float sc[NWIN][NWIN + 1];

  const int tid = threadIdx.x;
  for (int i = tid; i < NWIN * 32; i += 64) {
    int t = i >> 5, d = i & 31;
    size_t base = ((size_t)(w * NWIN + t)) * 576 + h * 32 + d;
    ks[t][d] = (float)qkvbr[base + 192];
    vs[t][d] = (float)qkvbr[base + 384] +
               (float)convbf[((size_t)(w * NWIN + t)) * CC + h * 32 + d];
  }
  if (tid < NWIN) {
    float s = 0.f;
    size_t base = ((size_t)(w * NWIN + tid)) * 576 + 192 + h * 32;
    for (int d = 0; d < 32; ++d) { float v = (float)qkvx[base + d]; s += v * v; }
    kn[tid] = sqrtf(s);
  }
  __syncthreads();
  if (tid >= NWIN) return;

  float q[32];
  float qn = 0.f;
  size_t qb = ((size_t)(w * NWIN + tid)) * 576 + h * 32;
#pragma unroll
  for (int d = 0; d < 32; ++d) { q[d] = (float)qkvx[qb + d] * SCALE_Q; qn += q[d] * q[d]; }
  qn = sqrtf(qn);

  const float* taup = tau + ((size_t)h * NWIN + tid) * NWIN;
  const float* rpbp = rpb + ((size_t)h * NWIN + tid) * NWIN;
  float mx = -1e30f;
  for (int j = 0; j < NWIN; ++j) {
    float dot = 0.f;
#pragma unroll
    for (int d = 0; d < 32; ++d) dot += q[d] * ks[j][d];
    float denom = fmaxf(qn * kn[j], 1e-6f);
    float tc = fmaxf(taup[j], 0.01f);
    float a = dot / denom / tc + rpbp[j];
    sc[tid][j] = a;
    mx = fmaxf(mx, a);
  }
  float sum = 0.f;
  for (int j = 0; j < NWIN; ++j) { float e = expf(sc[tid][j] - mx); sc[tid][j] = e; sum += e; }
  float inv = 1.f / sum;
  float o[32];
#pragma unroll
  for (int d = 0; d < 32; ++d) o[d] = 0.f;
  for (int j = 0; j < NWIN; ++j) {
    float p = sc[tid][j] * inv;
#pragma unroll
    for (int d = 0; d < 32; ++d) o[d] += p * vs[j][d];
  }
  __bf16* dst = cat + ((size_t)(w * NWIN + tid)) * 384 + br * CC + h * 32;
#pragma unroll
  for (int d = 0; d < 32; ++d) dst[d] = (__bf16)o[d];
}

// ---------------------------------------------------------------------------
// K7: window 3x3 conv as implicit WMMA GEMM. One block = one window,
// 128 threads = 4 waves; wave owns a 16-pixel M-tile x 192 cout.
// Input window async-staged into LDS (+zero row 49 for SAME padding).
// EPI 0: BN+GELU -> bf16.  EPI 1: BN+GELU + window-reverse + shortcut -> f32.
// ---------------------------------------------------------------------------
template <int CIN, int EPI>
__global__ __launch_bounds__(128) void conv_win_kernel(
    const __bf16* __restrict__ In, const __bf16* __restrict__ Wt,
    const float* __restrict__ cb, const float* __restrict__ g,
    const float* __restrict__ bb, const float* __restrict__ mn,
    const float* __restrict__ vr, __bf16* __restrict__ obf,
    const float* __restrict__ xres, float* __restrict__ of32) {
  constexpr int STR = CIN + 8;
  __shared__ __align__(16) __bf16 Is[50 * STR];

  const int tid = threadIdx.x;
  const int lane = tid & 31;
  const int mt = tid >> 5;
  const int lm = lane & 15;
  const int win = blockIdx.x;
  const size_t base = (size_t)win * NWIN * CIN;

  for (int i = tid * 8; i < NWIN * CIN; i += 128 * 8) {
    int r = i / CIN, c = i % CIN;
    async_g2l_b128(&Is[r * STR + c], In, (unsigned)((base + i) * 2));
  }
  for (int i = tid * 8; i < CIN; i += 128 * 8)
    *(uint4*)&Is[49 * STR + i] = make_uint4(0u, 0u, 0u, 0u);
  wait_async0();
  __syncthreads();

  const int mloc = mt * 16 + lm;            // output pixel this lane covers (A rows)
  v8f acc[12] = {};
  for (int tap = 0; tap < 9; ++tap) {
    int dy = tap / 3 - 1, dx = tap % 3 - 1;
    int py = mloc / 7 + dy, px = mloc % 7 + dx;
    bool ok = (mloc < NWIN) && (py >= 0) && (py < 7) && (px >= 0) && (px < 7);
    int sp = ok ? (py * 7 + px) : 49;       // row 49 is the zero row
    const __bf16* arow = &Is[sp * STR];
    for (int kc = 0; kc < CIN / 32; ++kc) {
      v16bf af = frag_a(arow + kc * 32, lane);
#pragma unroll
      for (int ni = 0; ni < 12; ++ni) {
        int co = ni * 16 + lm;
        const __bf16* brow = Wt + ((size_t)tap * CC + co) * CIN + kc * 32;
        v16bf bf = frag_b(brow, lane);
        acc[ni] = __builtin_amdgcn_wmma_f32_16x16x32_bf16(
            false, af, false, bf, (short)0, acc[ni], false, false);
      }
    }
  }

  const int hi8 = (lane >= 16) ? 8 : 0;
#pragma unroll
  for (int ni = 0; ni < 12; ++ni) {
    int co = ni * 16 + lm;
    float bias = cb[co];
    float s = g[co] * rsqrtf(vr[co] + 1e-5f);
    float mu = mn[co], beta = bb[co];
#pragma unroll
    for (int r = 0; r < 8; ++r) {
      int ml = mt * 16 + r + hi8;
      if (ml < NWIN) {
        float v = gelu_f((acc[ni][r] + bias - mu) * s + beta);
        if (EPI == 0) {
          obf[((size_t)win * NWIN + ml) * CC + co] = (__bf16)v;
        } else {
          int b = win >> 6;
          int wh = (win >> 3) & 7, ww = win & 7;
          int n = (wh * 7 + ml / 7) * 56 + (ww * 7 + ml % 7);
          size_t oi = ((size_t)b * 3136 + n) * CC + co;
          of32[oi] = xres[oi] + v;
        }
      }
    }
  }
}

// ---------------------------------------------------------------------------
// K8: LayerNorm over C=192 per row; one wave per row -> bf16
// ---------------------------------------------------------------------------
__global__ __launch_bounds__(256) void ln_kernel(
    const float* __restrict__ xo, const float* __restrict__ g,
    const float* __restrict__ b, __bf16* __restrict__ out) {
  int row = blockIdx.x * 8 + (threadIdx.x >> 5);
  int lane = threadIdx.x & 31;
  const float* px = xo + (size_t)row * CC;
  float v[6], s = 0.f, s2 = 0.f;
#pragma unroll
  for (int j = 0; j < 6; ++j) { v[j] = px[lane + j * 32]; s += v[j]; s2 += v[j] * v[j]; }
#pragma unroll
  for (int off = 16; off >= 1; off >>= 1) { s += __shfl_xor(s, off); s2 += __shfl_xor(s2, off); }
  float mu = s * (1.f / CC);
  float var = s2 * (1.f / CC) - mu * mu;
  float rstd = rsqrtf(var + 1e-5f);
  __bf16* po = out + (size_t)row * CC;
#pragma unroll
  for (int j = 0; j < 6; ++j) {
    int c = lane + j * 32;
    po[c] = (__bf16)(((v[j] - mu) * rstd) * g[c] + b[c]);
  }
}

// ---------------------------------------------------------------------------
extern "C" void kernel_launch(void* const* d_in, const int* in_sizes, int n_in,
                              void* d_out, int out_size, void* d_ws, size_t ws_size,
                              hipStream_t stream) {
  const float* x      = (const float*)d_in[0];
  const float* s1a    = (const float*)d_in[1];
  const float* s1d    = (const float*)d_in[2];
  const float* qkv_w  = (const float*)d_in[3];
  const float* qkv_b  = (const float*)d_in[4];
  const float* qkva_w = (const float*)d_in[5];
  const float* qkva_b = (const float*)d_in[6];
  const float* qkvd_w = (const float*)d_in[7];
  const float* qkvd_b = (const float*)d_in[8];
  const float* tau    = (const float*)d_in[9];
  const float* cpb_w1 = (const float*)d_in[10];
  const float* cpb_b1 = (const float*)d_in[11];
  const float* cpb_w2 = (const float*)d_in[12];
  const float* cpb_b2 = (const float*)d_in[13];
  const float* dw_w   = (const float*)d_in[14];
  const float* dw_b   = (const float*)d_in[15];
  const float* c1_w   = (const float*)d_in[16];
  const float* c1_b   = (const float*)d_in[17];
  const float* bn1_g  = (const float*)d_in[18];
  const float* bn1_b  = (const float*)d_in[19];
  const float* bn1_m  = (const float*)d_in[20];
  const float* bn1_v  = (const float*)d_in[21];
  const float* c2_w   = (const float*)d_in[22];
  const float* c2_b   = (const float*)d_in[23];
  const float* bn2_g  = (const float*)d_in[24];
  const float* bn2_b  = (const float*)d_in[25];
  const float* bn2_m  = (const float*)d_in[26];
  const float* bn2_v  = (const float*)d_in[27];
  const float* ln2_g  = (const float*)d_in[28];
  const float* ln2_b  = (const float*)d_in[29];
  const float* mlp_w1 = (const float*)d_in[30];
  const float* mlp_b1 = (const float*)d_in[31];
  const float* mlp_w2 = (const float*)d_in[32];
  const float* mlp_b2 = (const float*)d_in[33];
  (void)in_sizes; (void)n_in; (void)out_size; (void)ws_size;

  char* ws = (char*)d_ws;
  size_t off = 0;
  auto alloc = [&](size_t bytes) -> void* {
    void* p = ws + off;
    off += (bytes + 255) & ~(size_t)255;
    return p;
  };

  float*  rpb    = (float*)alloc((size_t)6 * 2401 * 4);
  __bf16* xw     = (__bf16*)alloc((size_t)TT * CC * 2);
  __bf16* aw     = (__bf16*)alloc((size_t)TT * CC * 2);
  __bf16* dwn    = (__bf16*)alloc((size_t)TT * CC * 2);
  __bf16* convbf = (__bf16*)alloc((size_t)TT * CC * 2);
  __bf16* qx     = (__bf16*)alloc((size_t)TT * 576 * 2);
  __bf16* qa     = (__bf16*)alloc((size_t)TT * 576 * 2);
  __bf16* qd     = (__bf16*)alloc((size_t)TT * 576 * 2);
  __bf16* cat    = (__bf16*)alloc((size_t)TT * 384 * 2);
  __bf16* y1     = (__bf16*)alloc((size_t)TT * CC * 2);
  float*  xo     = (float*)alloc((size_t)TT * CC * 4);
  __bf16* hln    = (__bf16*)alloc((size_t)TT * CC * 2);
  __bf16* hid    = (__bf16*)alloc((size_t)TT * 768 * 2);
  __bf16* wqx    = (__bf16*)alloc((size_t)576 * CC * 2);
  __bf16* wqa    = (__bf16*)alloc((size_t)576 * CC * 2);
  __bf16* wqd    = (__bf16*)alloc((size_t)576 * CC * 2);
  __bf16* wt1    = (__bf16*)alloc((size_t)9 * CC * 384 * 2);
  __bf16* wt2    = (__bf16*)alloc((size_t)9 * CC * CC * 2);
  __bf16* wm1    = (__bf16*)alloc((size_t)768 * CC * 2);
  __bf16* wm2    = (__bf16*)alloc((size_t)CC * 768 * 2);

  // --- small prep ---
  rpb_kernel<<<(2401 + 63) / 64, 64, 0, stream>>>(cpb_w1, cpb_b1, cpb_w2, cpb_b2, rpb);
  prep_kernel<<<(TT * CC) / 256, 256, 0, stream>>>(x, s1a, s1d, xw, aw, dwn);
  cvt_kernel<<<(576 * CC + 255) / 256, 256, 0, stream>>>(qkv_w,  wqx, 576 * CC);
  cvt_kernel<<<(576 * CC + 255) / 256, 256, 0, stream>>>(qkva_w, wqa, 576 * CC);
  cvt_kernel<<<(576 * CC + 255) / 256, 256, 0, stream>>>(qkvd_w, wqd, 576 * CC);
  cvt_kernel<<<(768 * CC + 255) / 256, 256, 0, stream>>>(mlp_w1, wm1, 768 * CC);
  cvt_kernel<<<(CC * 768 + 255) / 256, 256, 0, stream>>>(mlp_w2, wm2, CC * 768);
  repack_conv_kernel<384><<<(9 * CC * 384) / 256, 256, 0, stream>>>(c1_w, wt1);
  repack_conv_kernel<192><<<(9 * CC * 192) / 256, 256, 0, stream>>>(c2_w, wt2);
  dwconv_kernel<<<(TT * CC) / 256, 256, 0, stream>>>(xw, dw_w, dw_b, convbf);

  // --- QKV projections (bf16 WMMA GEMM, M=50176 K=192 N=576) ---
  dim3 gq(TT / 128, 576 / 64);
  gemm_kernel<0><<<gq, 256, 0, stream>>>(xw,  wqx, qkv_b,  (void*)qx, nullptr, TT, 576, CC);
  gemm_kernel<0><<<gq, 256, 0, stream>>>(aw,  wqa, qkva_b, (void*)qa, nullptr, TT, 576, CC);
  gemm_kernel<0><<<gq, 256, 0, stream>>>(dwn, wqd, qkvd_b, (void*)qd, nullptr, TT, 576, CC);

  // --- cosine attention, both branches -> cat ---
  attn_kernel<<<BW * 6 * 2, 64, 0, stream>>>(qx, qa, qd, convbf, tau, rpb, cat);

  // --- conv1 (2C->C) + BN + GELU, conv2 (C->C) + BN + GELU + reverse + shortcut ---
  conv_win_kernel<384, 0><<<BW, 128, 0, stream>>>(cat, wt1, c1_b, bn1_g, bn1_b, bn1_m,
                                                  bn1_v, y1, nullptr, nullptr);
  conv_win_kernel<192, 1><<<BW, 128, 0, stream>>>(y1, wt2, c2_b, bn2_g, bn2_b, bn2_m,
                                                  bn2_v, nullptr, x, xo);

  // --- LN + MLP (WMMA GEMMs), final residual into d_out ---
  ln_kernel<<<TT / 8, 256, 0, stream>>>(xo, ln2_g, ln2_b, hln);
  dim3 g1(TT / 128, 768 / 64);
  gemm_kernel<1><<<g1, 256, 0, stream>>>(hln, wm1, mlp_b1, (void*)hid, nullptr, TT, 768, CC);
  dim3 g2(TT / 128, CC / 64);
  gemm_kernel<2><<<g2, 256, 0, stream>>>(hid, wm2, mlp_b2, d_out, xo, TT, CC, 768);
}
